// SPEAttention_69406671503541
// MI455X (gfx1250) — matrix operand order repaired and checked by reading, not compiled, not run on hardware
//
#include <hip/hip_runtime.h>

#define Bb 32
#define Cc 256
#define Nn 4096  // 64*64
#define SROW 40  // padded LDS row stride in halfs (32 data + 8 pad)

typedef __bf16 bf16_t;
typedef __attribute__((ext_vector_type(16))) __bf16 v16bf;
typedef __attribute__((ext_vector_type(8)))  __bf16 v8bf;
typedef __attribute__((ext_vector_type(8)))  float  v8f;

#if __has_builtin(__builtin_amdgcn_tensor_load_to_lds)
#define USE_TDM 1
#else
#define USE_TDM 0
#endif

__device__ __forceinline__ bf16_t f2bf(float f) {
  unsigned u = __builtin_bit_cast(unsigned, f);
  unsigned r = (u + 0x7FFFu + ((u >> 16) & 1u)) >> 16;   // RNE
  unsigned short s = (unsigned short)r;
  return __builtin_bit_cast(bf16_t, s);
}

#define WMMA_BF16(A, B, C)                                                  \
  __builtin_amdgcn_wmma_f32_16x16x32_bf16(false, (A), false, (B), (short)0, \
                                          (C), false, false)

// load 16 halfs as a WMMA fragment: two 16B chunks, second at +step halfs
__device__ __forceinline__ v16bf ld32(const bf16_t* p, int step) {
  v8bf a0 = *(const v8bf*)p;
  v8bf a1 = *(const v8bf*)(p + step);
  return __builtin_shufflevector(a0, a1, 0, 1, 2, 3, 4, 5, 6, 7,
                                 8, 9, 10, 11, 12, 13, 14, 15);
}

// ---------------- depthwise 3x3 conv -> q,k (C x N bf16) and v^T (N x C bf16)
__global__ __launch_bounds__(256)
void dwconv_qkv(const float* __restrict__ x,
                const float* __restrict__ wq, const float* __restrict__ bq,
                const float* __restrict__ wk, const float* __restrict__ bk,
                const float* __restrict__ wv, const float* __restrict__ bv,
                bf16_t* __restrict__ qb, bf16_t* __restrict__ kb,
                bf16_t* __restrict__ vtb) {
  __shared__ float t[66 * 66];
  const int bc = blockIdx.x;            // b*C + c
  const int c = bc & (Cc - 1);
  const int b = bc >> 8;
  const int tid = threadIdx.x;
  const float* xp = x + (size_t)bc * Nn;
  for (int i = tid; i < 66 * 66; i += 256) {
    int hh = i / 66 - 1, ww = i % 66 - 1;
    float val = 0.f;
    if ((unsigned)hh < 64u && (unsigned)ww < 64u) val = xp[(hh << 6) + ww];
    t[i] = val;
  }
  __syncthreads();
  float fq[9], fk[9], fv[9];
#pragma unroll
  for (int j = 0; j < 9; ++j) {
    fq[j] = wq[c * 9 + j];
    fk[j] = wk[c * 9 + j];
    fv[j] = wv[c * 9 + j];
  }
  const float bqv = bq[c], bkv = bk[c], bvv = bv[c];
  for (int p = tid; p < Nn; p += 256) {
    const int h = p >> 6, w = p & 63;
    float aq = bqv, ak = bkv, av = bvv;
#pragma unroll
    for (int dy = 0; dy < 3; ++dy) {
      const float* row = &t[(h + dy) * 66 + w];
#pragma unroll
      for (int dx = 0; dx < 3; ++dx) {
        const float xv = row[dx];
        aq += xv * fq[dy * 3 + dx];
        ak += xv * fk[dy * 3 + dx];
        av += xv * fv[dy * 3 + dx];
      }
    }
    const size_t qi = (size_t)bc * Nn + p;
    qb[qi] = f2bf(aq);
    kb[qi] = f2bf(ak);
    vtb[((size_t)b * Nn + p) * Cc + c] = f2bf(av);   // transposed store
  }
}

__global__ __launch_bounds__(256)
void cvt_wp(const float* __restrict__ wp, bf16_t* __restrict__ wpb) {
  const int i = blockIdx.x * 256 + threadIdx.x;
  wpb[i] = f2bf(wp[i]);
}

// ---------------- GEMM1: attn = (q @ k^T) * 1/sqrt(C); 2x2 register blocking
__global__ __launch_bounds__(256)
void gemm_qk(const bf16_t* __restrict__ qb, const bf16_t* __restrict__ kb,
             float* __restrict__ attn) {
  const int wave = blockIdx.x * 8 + (threadIdx.x >> 5);
  const int lane = threadIdx.x & 31;
  const int b  = wave >> 6;          // 64 blocks of 32x32 per batch
  const int mB = (wave >> 3) & 7;
  const int dB = wave & 7;
  const int l15 = lane & 15;
  const bool hi = (lane & 16) != 0;
  const int aoff = hi ? 8 : 0, boff = hi ? 8 : 0;

  const bf16_t* a0 = qb + (size_t)(b * Cc + mB * 32 + l15) * Nn + aoff;
  const bf16_t* a1 = a0 + (size_t)16 * Nn;
  const bf16_t* b0 = kb + (size_t)(b * Cc + dB * 32 + l15) * Nn + boff;
  const bf16_t* b1 = b0 + (size_t)16 * Nn;

  v8f c00 = {}, c01 = {}, c10 = {}, c11 = {};
#pragma unroll 2
  for (int k0 = 0; k0 < Nn; k0 += 32) {
    __builtin_prefetch(a0 + k0 + 512, 0, 3);
    __builtin_prefetch(b0 + k0 + 512, 0, 3);
    v16bf A0 = ld32(a0 + k0, 16);
    v16bf A1 = ld32(a1 + k0, 16);
    v16bf B0 = ld32(b0 + k0, 16);
    v16bf B1 = ld32(b1 + k0, 16);
    c00 = WMMA_BF16(A0, B0, c00);
    c01 = WMMA_BF16(A0, B1, c01);
    c10 = WMMA_BF16(A1, B0, c10);
    c11 = WMMA_BF16(A1, B1, c11);
  }
  const float scale = 0.0625f;  // 256^-0.5
  float* ab = attn + (size_t)b * Cc * Cc;
  const int rhi = hi ? 8 : 0;
#pragma unroll
  for (int mi = 0; mi < 2; ++mi) {
#pragma unroll
    for (int ni = 0; ni < 2; ++ni) {
      const v8f acc = (mi == 0) ? (ni == 0 ? c00 : c01) : (ni == 0 ? c10 : c11);
      const int row0 = mB * 32 + mi * 16 + rhi;
      const int col = dB * 32 + ni * 16 + l15;
#pragma unroll
      for (int r = 0; r < 8; ++r)
        ab[(size_t)(row0 + r) * Cc + col] = acc[r] * scale;
    }
  }
}

// ---------------- row softmax over 256 elems, one wave per row, bf16 out
__global__ __launch_bounds__(256)
void softmax_rows(const float* __restrict__ attn, bf16_t* __restrict__ probs) {
  const int row = blockIdx.x * 8 + (threadIdx.x >> 5);
  const int lane = threadIdx.x & 31;
  const float* p = attn + (size_t)row * Cc;
  float v[8];
  float m = -3.4e38f;
#pragma unroll
  for (int j = 0; j < 8; ++j) {
    v[j] = p[lane + j * 32];
    m = fmaxf(m, v[j]);
  }
#pragma unroll
  for (int o = 16; o > 0; o >>= 1) m = fmaxf(m, __shfl_xor(m, o, 32));
  float s = 0.f;
#pragma unroll
  for (int j = 0; j < 8; ++j) {
    v[j] = __expf(v[j] - m);
    s += v[j];
  }
#pragma unroll
  for (int o = 16; o > 0; o >>= 1) s += __shfl_xor(s, o, 32);
  const float inv = 1.f / s;
  bf16_t* q = probs + (size_t)row * Cc;
#pragma unroll
  for (int j = 0; j < 8; ++j) q[lane + j * 32] = f2bf(v[j] * inv);
}

// =====================================================================
// Tensor Data Mover: DMA a 128-row x 32-half tile (row stride Cc halfs)
// into LDS with 16B padding every row (SROW=40 halfs), per ISA 08 section 8.
#if USE_TDM
typedef unsigned uint32x4 __attribute__((ext_vector_type(4)));
typedef int int32x8 __attribute__((ext_vector_type(8)));
typedef int int32x4 __attribute__((ext_vector_type(4)));

__device__ __forceinline__ void tdm_tile_load(unsigned lds_byte, const bf16_t* g) {
  const unsigned long long ga = (unsigned long long)g;
  uint32x4 g0;
  g0[0] = 1u;                                  // count=1 (valid user D#)
  g0[1] = lds_byte;                            // lds_addr
  g0[2] = (unsigned)ga;                        // global_addr[31:0]
  g0[3] = (unsigned)(ga >> 32) | (2u << 30);   // global_addr[56:32] | type=2
  int32x8 g1;
  // data_size=2B (1<<16), pad_enable (1<<20), pad_interval=3 (16 DW),
  // pad_amount=3 (4 DW) -> row stride 40 halfs in LDS
  g1[0] = (1 << 16) | (1 << 20) | (3 << 22) | (3 << 25);
  g1[1] = (int)(32u << 16);     // tensor_dim0 = 32 (tile-sized: no OOB)
  g1[2] = (int)(128u << 16);    // tensor_dim0_hi=0 | tensor_dim1 = 128
  g1[3] = (int)(32u << 16);     // tensor_dim1_hi=0 | tile_dim0 = 32
  g1[4] = 128;                  // tile_dim1 = 128, tile_dim2 = 0
  g1[5] = Cc;                   // tensor_dim0_stride = 256 elements
  g1[6] = 0;                    // stride hi / dim1_stride lo
  g1[7] = 0;
  int32x4 z4;
  z4[0] = 0; z4[1] = 0; z4[2] = 0; z4[3] = 0;
#if defined(__clang_major__) && (__clang_major__ >= 23)
  int32x8 z8;
  z8[0] = 0; z8[1] = 0; z8[2] = 0; z8[3] = 0;
  z8[4] = 0; z8[5] = 0; z8[6] = 0; z8[7] = 0;
  __builtin_amdgcn_tensor_load_to_lds(g0, g1, z4, z4, z8, 0);
#else
  __builtin_amdgcn_tensor_load_to_lds(g0, g1, z4, z4, 0);
#endif
}

__device__ __forceinline__ unsigned lds_off(const void* p) {
  return (unsigned)(unsigned long long)p;
}
#else
// fallback: cooperative synchronous staging (256 threads, 128x32 tile, padded)
__device__ __forceinline__ void stage_panel(bf16_t* s, const bf16_t* g, int tid) {
  const int row = tid >> 1;
  const int h = (tid & 1) * 16;
  const bf16_t* gp = g + (size_t)row * Cc + h;
  *(v8bf*)(s + row * SROW + h) = *(const v8bf*)gp;
  *(v8bf*)(s + row * SROW + h + 8) = *(const v8bf*)(gp + 8);
}
#endif

// =====================================================================
// LDS-tiled GEMM body: block computes 128x128 of D = A(128xK) * B^T? No --
// A rows are M (K contiguous), B rows are N (K contiguous). 8 waves 2x4,
// each wave 64x32 (4x2 WMMA tiles). K = 256, staged 32 per step, 2 buffers.
template <typename EPI>
__device__ __forceinline__ void lds_gemm_128x128(const bf16_t* gA, const bf16_t* gB,
                                                 EPI epilogue) {
  __shared__ bf16_t sA[2][128 * SROW];
  __shared__ bf16_t sB[2][128 * SROW];
  const int tid = threadIdx.x;
  const int warp = tid >> 5;
  const int lane = tid & 31;
  const int wm = warp >> 2;   // 0..1 -> M offset wm*64
  const int wn = warp & 3;    // 0..3 -> N offset wn*32
  const int l15 = lane & 15;
  const bool hi = (lane & 16) != 0;
  const int aoff = hi ? 8 : 0, boff = hi ? 8 : 0;

  v8f acc[4][2];
#pragma unroll
  for (int mi = 0; mi < 4; ++mi)
#pragma unroll
    for (int ni = 0; ni < 2; ++ni) acc[mi][ni] = {};

#if USE_TDM
  if (warp == 0) {
    tdm_tile_load(lds_off(&sA[0][0]), gA);
    tdm_tile_load(lds_off(&sB[0][0]), gB);
  }
#else
  stage_panel(&sA[0][0], gA, tid);
  stage_panel(&sB[0][0], gB, tid);
  __syncthreads();
#endif

  for (int s = 0; s < 8; ++s) {
    const int cur = s & 1;
#if USE_TDM
    if (warp == 0) {
      if (s < 7) {
        tdm_tile_load(lds_off(&sA[cur ^ 1][0]), gA + (s + 1) * 32);
        tdm_tile_load(lds_off(&sB[cur ^ 1][0]), gB + (s + 1) * 32);
        __builtin_amdgcn_s_wait_tensorcnt(2);  // current buffers complete
      } else {
        __builtin_amdgcn_s_wait_tensorcnt(0);
      }
    }
    __syncthreads();
#endif
    const bf16_t* As = &sA[cur][0];
    const bf16_t* Bs = &sB[cur][0];
    v16bf Af[4], Bf[2];
#pragma unroll
    for (int mi = 0; mi < 4; ++mi)
      Af[mi] = ld32(As + (wm * 64 + mi * 16 + l15) * SROW + aoff, 16);
#pragma unroll
    for (int ni = 0; ni < 2; ++ni)
      Bf[ni] = ld32(Bs + (wn * 32 + ni * 16 + l15) * SROW + boff, 16);
#pragma unroll
    for (int mi = 0; mi < 4; ++mi)
#pragma unroll
      for (int ni = 0; ni < 2; ++ni)
        acc[mi][ni] = WMMA_BF16(Af[mi], Bf[ni], acc[mi][ni]);
    __syncthreads();
#if !USE_TDM
    if (s < 7) {
      stage_panel(&sA[cur ^ 1][0], gA + (s + 1) * 32, tid);
      stage_panel(&sB[cur ^ 1][0], gB + (s + 1) * 32, tid);
    }
    __syncthreads();
#endif
  }
  epilogue(acc, wm, wn, l15, hi);
}

// ---------------- GEMM2: out^T = (probs @ v)^T -> bf16 (N x C), LDS tiled
__global__ __launch_bounds__(256)
void gemm_av(const bf16_t* __restrict__ probs, const bf16_t* __restrict__ vtb,
             bf16_t* __restrict__ outT) {
  const int blk = blockIdx.x;
  const int b = blk >> 6;
  const int bM = (blk >> 5) & 1;
  const int bN = blk & 31;
  const int M0 = bM * 128, N0 = bN * 128;
  const bf16_t* gA = probs + ((size_t)b * Cc + M0) * Cc;
  const bf16_t* gB = vtb + ((size_t)b * Nn + N0) * Cc;

  lds_gemm_128x128(gA, gB, [&](v8f (&acc)[4][2], int wm, int wn, int l15, bool hi) {
    const int rhi = hi ? 8 : 0;
#pragma unroll
    for (int mi = 0; mi < 4; ++mi)
#pragma unroll
      for (int ni = 0; ni < 2; ++ni) {
        const int n = N0 + wn * 32 + ni * 16 + l15;
        const int colBase = M0 + wm * 64 + mi * 16 + rhi;
        v8bf o;
#pragma unroll
        for (int r = 0; r < 8; ++r) o[r] = f2bf(acc[mi][ni][r]);
        *(v8bf*)(outT + ((size_t)b * Nn + n) * Cc + colBase) = o;
      }
  });
}

// ---------------- GEMM3: y = BN(wp @ out), f32 (B,C,H,W), LDS tiled
__global__ __launch_bounds__(256)
void gemm_proj_bn(const bf16_t* __restrict__ wpb, const bf16_t* __restrict__ outT,
                  const float* __restrict__ bn_w, const float* __restrict__ bn_b,
                  const float* __restrict__ bn_mean, const float* __restrict__ bn_var,
                  float* __restrict__ y) {
  const int blk = blockIdx.x;
  const int b = blk >> 6;
  const int bM = (blk >> 5) & 1;
  const int bN = blk & 31;
  const int M0 = bM * 128, N0 = bN * 128;
  const bf16_t* gA = wpb + (size_t)M0 * Cc;
  const bf16_t* gB = outT + ((size_t)b * Nn + N0) * Cc;

  lds_gemm_128x128(gA, gB, [&](v8f (&acc)[4][2], int wm, int wn, int l15, bool hi) {
    const int rhi = hi ? 8 : 0;
#pragma unroll
    for (int mi = 0; mi < 4; ++mi)
#pragma unroll
      for (int ni = 0; ni < 2; ++ni) {
        const int n = N0 + wn * 32 + ni * 16 + l15;
        const int oBase = M0 + wm * 64 + mi * 16 + rhi;
#pragma unroll
        for (int r = 0; r < 8; ++r) {
          const int o = oBase + r;
          const float inv = bn_w[o] * rsqrtf(bn_var[o] + 1e-5f);
          const float beta = bn_b[o] - bn_mean[o] * inv;
          y[(size_t)b * Cc * Nn + (size_t)o * Nn + n] = acc[mi][ni][r] * inv + beta;
        }
      }
  });
}

extern "C" void kernel_launch(void* const* d_in, const int* in_sizes, int n_in,
                              void* d_out, int out_size, void* d_ws, size_t ws_size,
                              hipStream_t stream) {
  (void)in_sizes; (void)n_in; (void)out_size; (void)ws_size;
  const float* x       = (const float*)d_in[0];
  const float* wq      = (const float*)d_in[1];
  const float* bq      = (const float*)d_in[2];
  const float* wk      = (const float*)d_in[3];
  const float* bk      = (const float*)d_in[4];
  const float* wv      = (const float*)d_in[5];
  const float* bv      = (const float*)d_in[6];
  const float* wp      = (const float*)d_in[7];
  const float* bn_w    = (const float*)d_in[8];
  const float* bn_b    = (const float*)d_in[9];
  const float* bn_mean = (const float*)d_in[10];
  const float* bn_var  = (const float*)d_in[11];
  float* y = (float*)d_out;

  char* ws = (char*)d_ws;
  bf16_t* qb    = (bf16_t*)(ws + (size_t)0);
  bf16_t* kb    = (bf16_t*)(ws + ((size_t)64 << 20));
  bf16_t* vtb   = (bf16_t*)(ws + ((size_t)128 << 20));
  float*  attn  = (float*)(ws + ((size_t)192 << 20));
  bf16_t* probs = (bf16_t*)(ws + ((size_t)200 << 20));
  bf16_t* outT  = (bf16_t*)(ws + ((size_t)204 << 20));
  bf16_t* wpb   = (bf16_t*)(ws + ((size_t)268 << 20));

  dwconv_qkv<<<Bb * Cc, 256, 0, stream>>>(x, wq, bq, wk, bk, wv, bv, qb, kb, vtb);
  cvt_wp<<<(Cc * Cc) / 256, 256, 0, stream>>>(wp, wpb);
  // GEMM1: 32 batches * 64 blocks of 32x32 = 2048 waves
  gemm_qk<<<256, 256, 0, stream>>>(qb, kb, attn);
  softmax_rows<<<1024, 256, 0, stream>>>(attn, probs);
  // GEMM2/3: 32 batches * 2 * 32 blocks of 128x128 = 2048 workgroups each
  gemm_av<<<2048, 256, 0, stream>>>(probs, vtb, outT);
  gemm_proj_bn<<<2048, 256, 0, stream>>>(wpb, outT, bn_w, bn_b, bn_mean, bn_var, y);
}
